// Attention_1675037245883
// MI455X (gfx1250) — compile-verified, matrix operand order checked by reading, not run on hardware
//
#include <hip/hip_runtime.h>
#include <hip/hip_bf16.h>

#define SEQLEN 2048
#define DMODEL 4096
#define NH 32
#define NKV 8
#define HD 128
#define RANK 16
#define KVD (NKV * HD) /* 1024 */

typedef __attribute__((ext_vector_type(16))) __bf16 v16bf;
typedef __attribute__((ext_vector_type(8))) float v8f;

union Frag16 {
    v16bf v;
    unsigned int u[8];
    unsigned short h[16];
};

__device__ __forceinline__ unsigned short f2bf(float f) {
    unsigned int u = __float_as_uint(f);
    u += 0x7FFFu + ((u >> 16) & 1u);   // round-to-nearest-even
    return (unsigned short)(u >> 16);
}
__device__ __forceinline__ float bf2f(unsigned short h) {
    return __uint_as_float(((unsigned int)h) << 16);
}
// Starting K-offset (elements) of the e8'th bf16-pair of a 16x32 A/B frag
// (CDNA5 ISA 7.12.2). Pairs are K-consecutive -> dword/b128 loads.
__device__ __forceinline__ int koff(int e8, int hlf) {
    return ((e8 >> 2) << 4) + (hlf << 3) + ((e8 & 3) << 1);
}

// ---- CDNA5 async memory->LDS (ASYNCcnt-tracked), GV addressing mode --------
__device__ __forceinline__ unsigned lds_off(const void* p) {
    // generic LDS pointer: low 32 bits are the LDS byte offset
    return (unsigned)(unsigned long long)p;
}
__device__ __forceinline__ void async_copy_b128(unsigned dst_lds, const void* src) {
    asm volatile("global_load_async_to_lds_b128 %0, %1, off"
                 :: "v"(dst_lds), "v"(src) : "memory");
}
__device__ __forceinline__ void wait_async() {
    asm volatile("s_wait_asynccnt 0x0" ::: "memory");
}

// ---------------------------------------------------------------- f32 -> bf16
__global__ void cvt_bf16_kernel(const float* __restrict__ X,
                                unsigned short* __restrict__ Y, int n) {
    int i = blockIdx.x * blockDim.x + threadIdx.x;
    int stride = gridDim.x * blockDim.x;
    for (; i < n; i += stride) Y[i] = f2bf(X[i]);
}

// ------------------------------------------------- LoRA A: T[s][r] = sum_d X[s][d]*A[r][d]
__global__ __launch_bounds__(256)
void lora_t_kernel(const unsigned short* __restrict__ Xb,
                   const float* __restrict__ A, float* __restrict__ T) {
    __shared__ float red[16][16];
    const int s = blockIdx.x;
    const int r = threadIdx.x & 15;
    const int part = threadIdx.x >> 4;
    const int base = part * (DMODEL / 16);
    float acc = 0.f;
    for (int i = 0; i < DMODEL / 16; ++i) {
        int d = base + i;
        acc += bf2f(Xb[(size_t)s * DMODEL + d]) * A[r * DMODEL + d];
    }
    red[part][r] = acc;
    __syncthreads();
    if (threadIdx.x < 16) {
        float t = 0.f;
#pragma unroll
        for (int p = 0; p < 16; ++p) t += red[p][threadIdx.x];
        T[(size_t)s * RANK + threadIdx.x] = t;
    }
}

// ---------------------------------------------------------------- WMMA GEMM
// C[s][o] = sum_k Abf[s][k]*Wbf[o][k]  (+ LoRA-B*T). MODE 0: RoPE + bf16 out,
// MODE 1: bf16 out, MODE 2: f32 out.
// Block = 128(S) x 64(O), 4 waves (each 32 S-rows x 64 O-cols = 8 WMMA tiles),
// K-step 32, double-buffered LDS filled by async-to-LDS b128 DMA.
template <int MODE>
__global__ __launch_bounds__(128)
void gemm_kernel(const unsigned short* __restrict__ Abf,
                 const unsigned short* __restrict__ Wbf,
                 const float* __restrict__ Blora,
                 const float* __restrict__ T,
                 void* __restrict__ Out, int O,
                 const float* __restrict__ FC, const float* __restrict__ FS) {
    __shared__ __align__(16) unsigned short As[2][128][32];  // 16 KB
    __shared__ __align__(16) unsigned short Ws[2][64][32];   //  8 KB
    __shared__ float Rt[4][16][16];                          //  4 KB

    const int tid = threadIdx.x;
    const int wid = tid >> 5, lane = tid & 31;
    const int hlf = lane >> 4, l15 = lane & 15;
    const int o0 = blockIdx.x * 64;
    const int s0 = blockIdx.y * 128;

    // per-thread staging coordinates (uint4 = 8 bf16)
    const int ar = tid >> 2, aq = tid & 3;     // A: rows tid>>2 + i*32
    const int wr = tid >> 2, wq2 = tid & 3;    // W: rows tid>>2 + i*32 (2 iters)

    auto issue_tile = [&](int kt, int buf) {
        const int k0 = kt * 32;
#pragma unroll
        for (int i = 0; i < 4; ++i) {  // A tile: 128x32 bf16 = 512 uint4
            int r = ar + i * 32;
            async_copy_b128(lds_off(&As[buf][r][aq * 8]),
                            Abf + (size_t)(s0 + r) * DMODEL + k0 + aq * 8);
        }
#pragma unroll
        for (int i = 0; i < 2; ++i) {  // W tile: 64x32 bf16 = 256 uint4
            int r = wr + i * 32;
            async_copy_b128(lds_off(&Ws[buf][r][wq2 * 8]),
                            Wbf + (size_t)(o0 + r) * DMODEL + k0 + wq2 * 8);
        }
    };

    v8f acc[2][4] = {};

    issue_tile(0, 0);
    wait_async();
    __syncthreads();

    const int NK = DMODEL / 32;
    for (int kt = 0; kt < NK; ++kt) {
        const int buf = kt & 1;
        if (kt + 1 < NK) issue_tile(kt + 1, buf ^ 1);

        Frag16 af[2];
#pragma unroll
        for (int mt = 0; mt < 2; ++mt)
#pragma unroll
            for (int e = 0; e < 8; ++e)
                af[mt].u[e] = *(const unsigned int*)
                    &As[buf][wid * 32 + mt * 16 + l15][koff(e, hlf)];
#pragma unroll
        for (int nt = 0; nt < 4; ++nt) {
            Frag16 b;
#pragma unroll
            for (int e = 0; e < 8; ++e)
                b.u[e] = *(const unsigned int*)&Ws[buf][nt * 16 + l15][koff(e, hlf)];
#pragma unroll
            for (int mt = 0; mt < 2; ++mt)
                acc[mt][nt] = __builtin_amdgcn_wmma_f32_16x16x32_bf16(
                    false, af[mt].v, false, b.v, (short)0, acc[mt][nt], false, false);
        }
        wait_async();
        __syncthreads();
    }

    // Epilogue: + LoRA-B * T, then optional RoPE, store.
#pragma unroll
    for (int nt = 0; nt < 4; ++nt) {
        const int o = o0 + nt * 16 + l15;
        float bl[RANK];
#pragma unroll
        for (int r = 0; r < RANK; ++r) bl[r] = Blora[(size_t)o * RANK + r];
#pragma unroll
        for (int mt = 0; mt < 2; ++mt) {
#pragma unroll
            for (int j = 0; j < 8; ++j) {
                const int s = s0 + wid * 32 + mt * 16 + j + 8 * hlf;
                float c = acc[mt][nt][j];
                float add = 0.f;
#pragma unroll
                for (int r = 0; r < RANK; ++r) add += bl[r] * T[(size_t)s * RANK + r];
                c += add;
                if (MODE == 2) {
                    ((float*)Out)[(size_t)s * O + o] = c;
                } else if (MODE == 1) {
                    ((unsigned short*)Out)[(size_t)s * O + o] = f2bf(c);
                } else {
                    Rt[wid][j + 8 * hlf][l15] = c;
                }
            }
            if (MODE == 0) {  // RoPE pairs live in adjacent lanes -> via LDS
                __syncthreads();
#pragma unroll
                for (int pi = 0; pi < 4; ++pi) {
                    int p = lane * 4 + pi;  // 128 pairs per 16x16 tile
                    int r = p >> 3, cp = p & 7;
                    int c0 = cp * 2;
                    int srow = s0 + wid * 32 + mt * 16 + r;
                    int oc = o0 + nt * 16 + c0;
                    int fi = (oc & (HD - 1)) >> 1;
                    float cs = FC[srow * (HD / 2) + fi];
                    float sn = FS[srow * (HD / 2) + fi];
                    float xe = Rt[wid][r][c0], xo = Rt[wid][r][c0 + 1];
                    ((unsigned short*)Out)[(size_t)srow * O + oc]     = f2bf(xe * cs - xo * sn);
                    ((unsigned short*)Out)[(size_t)srow * O + oc + 1] = f2bf(xe * sn + xo * cs);
                }
                __syncthreads();
            }
        }
    }
}

// ------------------------------------------------------------ flash attention
// One workgroup = head h, 64 query rows (4 waves x 16). Causal, GQA 4:1.
__global__ __launch_bounds__(128)
void attn_kernel(const unsigned short* __restrict__ Q,
                 const unsigned short* __restrict__ K,
                 const unsigned short* __restrict__ V,
                 unsigned short* __restrict__ Oout) {
    __shared__ __align__(16) unsigned short Kt[32][128];   // [key][hd]
    __shared__ __align__(16) unsigned short VtT[128][32];  // [hd][key]
    __shared__ __align__(4)  unsigned short Pt[4][16][32]; // per-wave P tile

    const int tid = threadIdx.x;
    const int wid = tid >> 5, lane = tid & 31;
    const int hlf = lane >> 4, l15 = lane & 15;
    const int q0 = blockIdx.x * 64;
    const int h = blockIdx.y;
    const int kvh = h >> 2;
    const float scale = 0.08838834764831845f;  // 1/sqrt(128)

    Frag16 qf[4];
    {
        const int qrow = q0 + wid * 16 + l15;
        const unsigned short* qp = Q + (size_t)qrow * DMODEL + h * HD;
#pragma unroll
        for (int kc = 0; kc < 4; ++kc)
#pragma unroll
            for (int e = 0; e < 8; ++e)
                qf[kc].u[e] = *(const unsigned int*)(qp + kc * 32 + koff(e, hlf));
    }

    v8f oacc[8] = {};
    float mrow[8], lrow[8];
#pragma unroll
    for (int j = 0; j < 8; ++j) { mrow[j] = -1e30f; lrow[j] = 0.f; }

    const int nkb = q0 / 32 + 2;  // keys [0, q0+64) cover causal span
    for (int kb = 0; kb < nkb; ++kb) {
        const int kbase = kb * 32;
#pragma unroll
        for (int i = 0; i < 4; ++i) {  // stage K block via async DMA
            int idx = tid + i * 128;
            int r = idx >> 4, c = idx & 15;
            async_copy_b128(lds_off(&Kt[r][c * 8]),
                            K + (size_t)(kbase + r) * KVD + kvh * HD + c * 8);
        }
#pragma unroll
        for (int i = 0; i < 32; ++i) {  // stage V block transposed (manual)
            int idx = tid + i * 128;
            int r = idx >> 7, c = idx & 127;
            VtT[c][r] = V[(size_t)(kbase + r) * KVD + kvh * HD + c];
        }
        wait_async();
        __syncthreads();

        v8f sf[2] = {};
#pragma unroll
        for (int g = 0; g < 2; ++g)
#pragma unroll
            for (int kc = 0; kc < 4; ++kc) {
                Frag16 b;
#pragma unroll
                for (int e = 0; e < 8; ++e)
                    b.u[e] = *(const unsigned int*)&Kt[g * 16 + l15][kc * 32 + koff(e, hlf)];
                sf[g] = __builtin_amdgcn_wmma_f32_16x16x32_bf16(
                    false, qf[kc].v, false, b.v, (short)0, sf[g], false, false);
            }

        const int key0 = kbase + l15;
        const int key1 = kbase + 16 + l15;
        float alpha[8];
#pragma unroll
        for (int j = 0; j < 8; ++j) {
            const int row = q0 + wid * 16 + j + 8 * hlf;
            float s0v = sf[0][j] * scale; if (key0 > row) s0v = -1e9f;
            float s1v = sf[1][j] * scale; if (key1 > row) s1v = -1e9f;
            float mx = fmaxf(s0v, s1v);
#pragma unroll
            for (int off = 1; off < 16; off <<= 1)
                mx = fmaxf(mx, __shfl_xor(mx, off, 32));
            float mnew = fmaxf(mrow[j], mx);
            float a = __expf(mrow[j] - mnew);
            float p0 = __expf(s0v - mnew);
            float p1 = __expf(s1v - mnew);
            float rs = p0 + p1;
#pragma unroll
            for (int off = 1; off < 16; off <<= 1) rs += __shfl_xor(rs, off, 32);
            lrow[j] = lrow[j] * a + rs;
            mrow[j] = mnew;
            alpha[j] = a;
            Pt[wid][j + 8 * hlf][l15]      = f2bf(p0);
            Pt[wid][j + 8 * hlf][16 + l15] = f2bf(p1);
        }
#pragma unroll
        for (int nt = 0; nt < 8; ++nt)
#pragma unroll
            for (int j = 0; j < 8; ++j) oacc[nt][j] *= alpha[j];
        __syncthreads();

        Frag16 pa;  // P as 16x32 bf16 A-fragment
#pragma unroll
        for (int e = 0; e < 8; ++e)
            pa.u[e] = *(const unsigned int*)&Pt[wid][l15][koff(e, hlf)];
#pragma unroll
        for (int nt = 0; nt < 8; ++nt) {  // O += P * V
            Frag16 b;
#pragma unroll
            for (int e = 0; e < 8; ++e)
                b.u[e] = *(const unsigned int*)&VtT[nt * 16 + l15][koff(e, hlf)];
            oacc[nt] = __builtin_amdgcn_wmma_f32_16x16x32_bf16(
                false, pa.v, false, b.v, (short)0, oacc[nt], false, false);
        }
        __syncthreads();
    }

#pragma unroll
    for (int nt = 0; nt < 8; ++nt)
#pragma unroll
        for (int j = 0; j < 8; ++j) {
            const int row = q0 + wid * 16 + j + 8 * hlf;
            float o = oacc[nt][j] / lrow[j];
            Oout[(size_t)row * DMODEL + h * HD + nt * 16 + l15] = f2bf(o);
        }
}

// ---------------------------------------------------------------------- launch
extern "C" void kernel_launch(void* const* d_in, const int* in_sizes, int n_in,
                              void* d_out, int out_size, void* d_ws, size_t ws_size,
                              hipStream_t stream) {
    (void)in_sizes; (void)n_in; (void)out_size; (void)ws_size;
    const float* x   = (const float*)d_in[0];
    const float* fc  = (const float*)d_in[2];
    const float* fs  = (const float*)d_in[3];
    const float* wq  = (const float*)d_in[5];
    const float* wk  = (const float*)d_in[6];
    const float* wv  = (const float*)d_in[7];
    const float* wo  = (const float*)d_in[8];
    const float* lqA = (const float*)d_in[9];
    const float* lqB = (const float*)d_in[10];
    const float* lkA = (const float*)d_in[11];
    const float* lkB = (const float*)d_in[12];
    const float* lvA = (const float*)d_in[13];
    const float* lvB = (const float*)d_in[14];
    const float* loA = (const float*)d_in[15];
    const float* loB = (const float*)d_in[16];

    char* ws = (char*)d_ws;
    unsigned short* xb  = (unsigned short*)(ws);                   // 16 MB
    unsigned short* qb  = (unsigned short*)(ws + (16ull << 20));   // 16 MB
    unsigned short* kbf = (unsigned short*)(ws + (32ull << 20));   //  4 MB
    unsigned short* vbf = (unsigned short*)(ws + (36ull << 20));   //  4 MB
    unsigned short* ab  = (unsigned short*)(ws + (40ull << 20));   // 16 MB
    unsigned short* wqb = (unsigned short*)(ws + (56ull << 20));   // 32 MB
    unsigned short* wkb = (unsigned short*)(ws + (88ull << 20));   //  8 MB
    unsigned short* wvb = (unsigned short*)(ws + (96ull << 20));   //  8 MB
    unsigned short* wob = (unsigned short*)(ws + (104ull << 20));  // 32 MB
    float* tq = (float*)(ws + (136ull << 20));                     // 128 KB each
    float* tk = (float*)(ws + (136ull << 20) + (1ull << 19));
    float* tv = (float*)(ws + (136ull << 20) + (2ull << 19));
    float* to = (float*)(ws + (136ull << 20) + (3ull << 19));

    // one-time bf16 conversions (bandwidth-bound, ~10us total at 23.3 TB/s)
    cvt_bf16_kernel<<<4096, 256, 0, stream>>>(x,  xb,  SEQLEN * DMODEL);
    cvt_bf16_kernel<<<4096, 256, 0, stream>>>(wq, wqb, DMODEL * DMODEL);
    cvt_bf16_kernel<<<4096, 256, 0, stream>>>(wk, wkb, KVD * DMODEL);
    cvt_bf16_kernel<<<4096, 256, 0, stream>>>(wv, wvb, KVD * DMODEL);
    cvt_bf16_kernel<<<4096, 256, 0, stream>>>(wo, wob, DMODEL * DMODEL);

    lora_t_kernel<<<SEQLEN, 256, 0, stream>>>(xb, lqA, tq);
    lora_t_kernel<<<SEQLEN, 256, 0, stream>>>(xb, lkA, tk);
    lora_t_kernel<<<SEQLEN, 256, 0, stream>>>(xb, lvA, tv);

    gemm_kernel<0><<<dim3(DMODEL / 64, SEQLEN / 128), 128, 0, stream>>>(
        xb, wqb, lqB, tq, qb, DMODEL, fc, fs);
    gemm_kernel<0><<<dim3(KVD / 64, SEQLEN / 128), 128, 0, stream>>>(
        xb, wkb, lkB, tk, kbf, KVD, fc, fs);
    gemm_kernel<1><<<dim3(KVD / 64, SEQLEN / 128), 128, 0, stream>>>(
        xb, wvb, lvB, tv, vbf, KVD, nullptr, nullptr);

    attn_kernel<<<dim3(SEQLEN / 64, NH), 128, 0, stream>>>(qb, kbf, vbf, ab);

    lora_t_kernel<<<SEQLEN, 256, 0, stream>>>(ab, loA, to);
    gemm_kernel<2><<<dim3(DMODEL / 64, SEQLEN / 128), 128, 0, stream>>>(
        ab, wob, loB, to, d_out, DMODEL, nullptr, nullptr);
}
